// GCModule_13546326852231
// MI455X (gfx1250) — matrix operand order repaired and checked by reading, not compiled, not run on hardware
//
#include <hip/hip_runtime.h>
#include <hip/hip_bf16.h>
#include <math.h>

typedef __attribute__((ext_vector_type(16))) _Float16 v16h;
typedef __attribute__((ext_vector_type(8)))  _Float16 v8h;
typedef __attribute__((ext_vector_type(8)))  float    v8f;

#define BATCH 4
#define C_IN 128
#define C_MID 64
#define HH 48
#define WW 48
#define HW (HH*WW)          // 2304
#define H2 (HH*2)           // 96
#define W2 (WW*2)           // 96
#define NENC 100
#define NENC_PAD 112
#define KENC 576            // 64 * 9, reordered as (kh*3+kw)*64 + ci
#define PADW 50             // comp output padded 50x50

// ---- workspace layout (bytes, all 256-aligned) ----
#define OFF_GAMMA   0u         // 4 f32
#define OFF_CSCALE  256u       // 64 f32
#define OFF_CBIAS   512u       // 64 f32
#define OFF_ESCALE  768u       // 112 f32
#define OFF_EBIAS   1280u      // 112 f32
#define OFF_COMPW   2048u      // 64*128 f16  = 16384 B
#define OFF_ENCW    18432u     // 112*576 f16 = 129024 B
#define OFF_XGH     147456u    // 4*2304*128 f16 (channel-last) = 2359296 B
#define OFF_XGF     2506752u   // 4*2304*128 f32 (channel-last) = 4718592 B
#define OFF_CPAD    7225344u   // 4*50*50*64 f16 = 1280000 B
#define OFF_ENC     8505344u   // 4*100*2304 f32 = 3686400 B

// ------------------------------------------------------------------
// 1) weight prep: BN folding, f16 conversion, enc-weight K reorder,
//    and zeroing of the padded comp buffer (borders must be 0 each call).
// ------------------------------------------------------------------
__global__ __launch_bounds__(256)
void k_prep(const float* __restrict__ comp_w,
            const float* __restrict__ cg, const float* __restrict__ cbn,
            const float* __restrict__ cm, const float* __restrict__ cv,
            const float* __restrict__ enc_w,
            const float* __restrict__ eg, const float* __restrict__ ebn,
            const float* __restrict__ em, const float* __restrict__ ev,
            unsigned char* __restrict__ ws) {
  int tid = blockIdx.x * 256 + threadIdx.x;     // grid covers 320000 threads
  _Float16* compw = (_Float16*)(ws + OFF_COMPW);
  _Float16* encw  = (_Float16*)(ws + OFF_ENCW);
  float* cs  = (float*)(ws + OFF_CSCALE);
  float* cbs = (float*)(ws + OFF_CBIAS);
  float* es  = (float*)(ws + OFF_ESCALE);
  float* ebs = (float*)(ws + OFF_EBIAS);
  unsigned int* cpadw = (unsigned int*)(ws + OFF_CPAD);

  if (tid < NENC_PAD * KENC) {
    int n = tid / KENC, k = tid % KENC;
    int s = k >> 6, ci = k & 63;                // k' = s*64 + ci, s = kh*3+kw
    float v = 0.f;
    if (n < NENC) v = enc_w[((n * 64 + ci) * 3 + s / 3) * 3 + (s % 3)];
    encw[tid] = (_Float16)v;
  }
  if (tid < C_MID * C_IN) compw[tid] = (_Float16)comp_w[tid];
  if (tid < C_MID) {
    float s = cg[tid] * rsqrtf(cv[tid] + 1e-3f);
    cs[tid] = s; cbs[tid] = cbn[tid] - cm[tid] * s;
  }
  if (tid < NENC_PAD) {
    if (tid < NENC) {
      float s = eg[tid] * rsqrtf(ev[tid] + 1e-3f);
      es[tid] = s; ebs[tid] = ebn[tid] - em[tid] * s;
    } else { es[tid] = 0.f; ebs[tid] = 0.f; }
  }
  if (tid < (BATCH * PADW * PADW * C_MID) / 2) cpadw[tid] = 0u;  // 320000 words
}

// ------------------------------------------------------------------
// 2) per-batch gray stats -> gamma  (ddof=1 std, as torch)
// ------------------------------------------------------------------
__global__ __launch_bounds__(256)
void k_gamma(const float* __restrict__ x, float* __restrict__ gamma) {
  __shared__ float s1[256], s2[256];
  int b = blockIdx.x, t = threadIdx.x;
  const float* p0 = x + (size_t)b * C_IN * HW;
  float sum = 0.f, sq = 0.f;
  for (int i = t; i < HW; i += 256) {
    float g = 0.299f * p0[i] + 0.587f * p0[HW + i] + 0.114f * p0[2 * HW + i];
    sum += g; sq += g * g;
  }
  s1[t] = sum; s2[t] = sq;
  __syncthreads();
  for (int off = 128; off > 0; off >>= 1) {
    if (t < off) { s1[t] += s1[t + off]; s2[t] += s2[t + off]; }
    __syncthreads();
  }
  if (t == 0) {
    float n = (float)HW;
    float mean = s1[0] / n;
    float var = (s2[0] - s1[0] * s1[0] / n) / (n - 1.f);
    float sd = sqrtf(fmaxf(var, 0.f));
    float g = 0.8f + 0.4f * mean / (mean + fmaxf(sd, 1e-6f));
    gamma[b] = fminf(fmaxf(g, 0.8f), 1.2f);
  }
}

// ------------------------------------------------------------------
// 3) x_g = (x + eps)^gamma  -> channel-last f32 (CARAFE) + f16 (GEMM A)
//    base is strictly positive and ~[1e-6, 1], gamma in [0.8,1.2]:
//    native exp2(gamma*log2(base)) is accurate to ~1 ulp*|log2| here.
// ------------------------------------------------------------------
__global__ __launch_bounds__(256)
void k_pow(const float* __restrict__ x, const float* __restrict__ gamma,
           float* __restrict__ xgf, _Float16* __restrict__ xgh) {
  int idx = blockIdx.x * 256 + threadIdx.x;     // 4*128*2304 = 1179648 exact
  int b = idx / (C_IN * HW);
  int c = (idx / HW) % C_IN;
  int pix = idx % HW;
  float v = __powf(x[idx] + 1e-6f, gamma[b]);
  int o = (b * HW + pix) * C_IN + c;
  xgf[o] = v;
  xgh[o] = (_Float16)v;
}

// ---- WMMA fragment helpers -----------------------------------------
// A (16x32 f16): lane m = l&15, kb = (l<16 ? 0 : 8); halves 0-7 = K kb..kb+7,
//                halves 8-15 = K kb+16..kb+23  -> two contiguous 16B loads.
// B (32x16 f16, stored as B^T [N][K]): lane n = l&15, kb = (l<16 ? 0 : 8);
//                halves 0-7 = K kb..kb+7, halves 8-15 = K kb+16..kb+23.
__device__ __forceinline__ v16h frag_cat(v8h lo, v8h hi) {
  v16h r;
#pragma unroll
  for (int e = 0; e < 8; ++e) { r[e] = lo[e]; r[e + 8] = hi[e]; }
  return r;
}

// fast sigmoid: denominator in (1, inf) -> v_rcp_f32 is well conditioned
__device__ __forceinline__ float fast_silu(float v) {
  return v * __builtin_amdgcn_rcpf(1.f + __expf(-v));
}

// ------------------------------------------------------------------
// 4) comp 1x1 conv as WMMA GEMM: M=9216 (pixels), N=64, K=128
//    epilogue: BN + SiLU -> padded channel-last f16 buffer
// ------------------------------------------------------------------
__global__ __launch_bounds__(256)
void k_comp_gemm(const _Float16* __restrict__ xgh,
                 const _Float16* __restrict__ compw,
                 const float* __restrict__ cs, const float* __restrict__ cbias,
                 _Float16* __restrict__ cpad) {
  int wid  = blockIdx.x * 8 + (threadIdx.x >> 5);   // 2304 waves exact
  int lane = threadIdx.x & 31;
  int mt = wid >> 2, nt = wid & 3;
  int hi = lane >> 4;
  int mrow = mt * 16 + (lane & 15);
  int n    = nt * 16 + (lane & 15);
  const _Float16* Ap = xgh + mrow * C_IN + (hi << 3);   // +kb
  const _Float16* Bp = compw + n * C_IN + (hi << 3);    // +kb
  v8f acc = {0.f, 0.f, 0.f, 0.f, 0.f, 0.f, 0.f, 0.f};
#pragma unroll
  for (int kc = 0; kc < 4; ++kc) {
    int k0 = kc * 32;
    v16h af = frag_cat(*(const v8h*)(Ap + k0), *(const v8h*)(Ap + k0 + 16));
    v16h bf = frag_cat(*(const v8h*)(Bp + k0), *(const v8h*)(Bp + k0 + 16));
    acc = __builtin_amdgcn_wmma_f32_16x16x32_f16(false, af, false, bf,
                                                 (short)0, acc, false, false);
  }
  // all 16 rows of the tile live in one image row: hoist b/y/x computation
  int mbase = mt * 16;
  int b0 = mbase / HW, rem0 = mbase % HW;
  int y0 = rem0 / WW, x0 = rem0 % WW;
  float sc = cs[n], bi = cbias[n];
  _Float16* outp = cpad + ((b0 * PADW + y0 + 1) * PADW + (x0 + 1)) * C_MID + n;
#pragma unroll
  for (int r = 0; r < 8; ++r) {           // C/D: lane col n, rows r / r+8
    int ml = r + (hi << 3);
    outp[ml * C_MID] = (_Float16)fast_silu(acc[r] * sc + bi);
  }
}

// ------------------------------------------------------------------
// 5) enc 3x3 conv as WMMA GEMM: per-batch M=2304, N=112(pad), K=576.
//    The whole B matrix (112x576 f16 = 129 KB) is staged into LDS with
//    gfx1250 async-to-LDS copies; each wave then computes 16 pixels x
//    all 7 N-tiles (register blocking: 1 A-fragment feeds 7 WMMAs).
// ------------------------------------------------------------------
__global__ __launch_bounds__(256)
void k_enc_gemm(const _Float16* __restrict__ cpad,
                const _Float16* __restrict__ encw,
                const float* __restrict__ es, const float* __restrict__ ebias,
                float* __restrict__ enc_out) {
  __shared__ _Float16 sB[NENC_PAD * KENC];          // 129024 B of LDS
  int t = threadIdx.x;

  // ---- stage B into LDS via async copies (ASYNCcnt path) ----
  {
    unsigned ldsBase = (unsigned)(size_t)(void*)&sB[0];
    for (int i = t; i < (NENC_PAD * KENC) / 8; i += 256) {   // 8064 x 16B
      unsigned laddr = ldsBase + (unsigned)i * 16u;
      unsigned long long gaddr = (unsigned long long)(const void*)(encw + i * 8);
      asm volatile("global_load_async_to_lds_b128 %0, %1, off"
                   :: "v"(laddr), "v"(gaddr) : "memory");
    }
    asm volatile("s_wait_asynccnt 0x0" ::: "memory");
    __syncthreads();
  }

  int lane = t & 31;
  int mt = blockIdx.x * 8 + (t >> 5);               // 576 mtiles = 72 blocks
  int b = mt / 144, rem = mt % 144;
  int y = rem / 3, xseg = (rem % 3) * 16;
  int hi = lane >> 4;
  int x = xseg + (lane & 15);
  int kb = hi << 3;
  int nb = lane & 15;
  const _Float16* sBl = &sB[nb * KENC + (hi << 3)]; // lane's B row base

  v8f acc[7];
#pragma unroll
  for (int nt = 0; nt < 7; ++nt)
    acc[nt] = (v8f){0.f, 0.f, 0.f, 0.f, 0.f, 0.f, 0.f, 0.f};

#pragma unroll
  for (int kc = 0; kc < 18; ++kc) {
    int k0 = kc * 32;
    int s = k0 >> 6, c0 = k0 & 63;        // tap index, channel base (0/32)
    int dy = s / 3, dx = s % 3;           // padded offsets (pad=1 folded in)
    const _Float16* Ap =
        cpad + ((b * PADW + y + dy) * PADW + (x + dx)) * C_MID + c0 + kb;
    if (kc + 1 < 18) {                    // prefetch next A chunk
      int k1 = k0 + 32;
      int s1 = k1 >> 6, c1 = k1 & 63;
      __builtin_prefetch(cpad + ((b * PADW + y + s1 / 3) * PADW +
                                 (x + s1 % 3)) * C_MID + c1 + kb, 0, 1);
    }
    v16h af = frag_cat(*(const v8h*)(Ap), *(const v8h*)(Ap + 16));
#pragma unroll
    for (int nt = 0; nt < 7; ++nt) {      // 7 WMMAs share one A fragment
      const _Float16* Bp = sBl + nt * 16 * KENC + k0;
      v16h bf = frag_cat(*(const v8h*)(Bp), *(const v8h*)(Bp + 16));
      acc[nt] = __builtin_amdgcn_wmma_f32_16x16x32_f16(false, af, false, bf,
                                                       (short)0, acc[nt],
                                                       false, false);
    }
  }

#pragma unroll
  for (int nt = 0; nt < 7; ++nt) {
    int n = nt * 16 + nb;
    if (n < NENC) {
      float sc = es[n], bi = ebias[n];
      float* op = enc_out + ((b * NENC + n) * HH + y) * WW + xseg;
#pragma unroll
      for (int r = 0; r < 8; ++r)
        op[r + (hi << 3)] = acc[nt][r] * sc + bi;
    }
  }
}

// ------------------------------------------------------------------
// 6) CARAFE: pixel-shuffle + softmax(25) + 5x5 dil-2 reassembly.
//    One wave per upsampled pixel; lane handles 4 channels (32*4 = 128).
// ------------------------------------------------------------------
__global__ __launch_bounds__(256)
void k_carafe(const float* __restrict__ enc_out,
              const float* __restrict__ xgf,
              float* __restrict__ out) {
  int wid  = blockIdx.x * 8 + (threadIdx.x >> 5);   // 36864 waves exact
  int lane = threadIdx.x & 31;
  int b = wid / (H2 * W2);
  int rem = wid % (H2 * W2);
  int Y = rem / W2, X = rem % W2;
  int y = Y >> 1, x = X >> 1;
  int sub = (Y & 1) * 2 + (X & 1);                  // pixel-shuffle sub-index
  const float* ep = enc_out + (size_t)b * NENC * HW + (size_t)y * WW + x;
  float wv[25];
  float mx = -3.4e38f;
#pragma unroll
  for (int kk = 0; kk < 25; ++kk) {
    wv[kk] = ep[(size_t)(kk * 4 + sub) * HW];
    mx = fmaxf(mx, wv[kk]);
  }
  float ssum = 0.f;
#pragma unroll
  for (int kk = 0; kk < 25; ++kk) { wv[kk] = __expf(wv[kk] - mx); ssum += wv[kk]; }
  float inv = __builtin_amdgcn_rcpf(ssum);   // ssum >= 1, well conditioned
#pragma unroll
  for (int kk = 0; kk < 25; ++kk) wv[kk] *= inv;

  float a0 = 0.f, a1 = 0.f, a2 = 0.f, a3 = 0.f;
  int cb = lane * 4;
#pragma unroll
  for (int i = 0; i < 5; ++i) {
    int yy = Y + 2 * i - 4;
    if (yy < 0 || yy >= H2) continue;
    int ys = yy >> 1;                                // nearest-upsample source
#pragma unroll
    for (int j = 0; j < 5; ++j) {
      int xx = X + 2 * j - 4;
      if (xx < 0 || xx >= W2) continue;
      int xs = xx >> 1;
      const float4 v =
          *(const float4*)(xgf + (size_t)((b * HH + ys) * WW + xs) * C_IN + cb);
      float w = wv[i * 5 + j];
      a0 += w * v.x; a1 += w * v.y; a2 += w * v.z; a3 += w * v.w;
    }
  }
  size_t ob = ((size_t)(b * C_IN + cb) * H2 + Y) * W2 + X;
  out[ob] = a0;
  out[ob + (size_t)H2 * W2]     = a1;
  out[ob + 2 * (size_t)H2 * W2] = a2;
  out[ob + 3 * (size_t)H2 * W2] = a3;
}

// ------------------------------------------------------------------
extern "C" void kernel_launch(void* const* d_in, const int* in_sizes, int n_in,
                              void* d_out, int out_size, void* d_ws, size_t ws_size,
                              hipStream_t stream) {
  (void)in_sizes; (void)n_in; (void)out_size; (void)ws_size;
  const float* x      = (const float*)d_in[0];
  const float* comp_w = (const float*)d_in[1];
  const float* cg     = (const float*)d_in[2];
  const float* cbn    = (const float*)d_in[3];
  const float* cm     = (const float*)d_in[4];
  const float* cv     = (const float*)d_in[5];
  const float* enc_w  = (const float*)d_in[6];
  const float* eg     = (const float*)d_in[7];
  const float* ebn    = (const float*)d_in[8];
  const float* em     = (const float*)d_in[9];
  const float* ev     = (const float*)d_in[10];
  unsigned char* ws = (unsigned char*)d_ws;
  float* out = (float*)d_out;

  float*     gamma = (float*)(ws + OFF_GAMMA);
  float*     cs    = (float*)(ws + OFF_CSCALE);
  float*     cbias = (float*)(ws + OFF_CBIAS);
  float*     esc   = (float*)(ws + OFF_ESCALE);
  float*     ebias = (float*)(ws + OFF_EBIAS);
  _Float16*  compw = (_Float16*)(ws + OFF_COMPW);
  _Float16*  encw  = (_Float16*)(ws + OFF_ENCW);
  _Float16*  xgh   = (_Float16*)(ws + OFF_XGH);
  float*     xgf   = (float*)(ws + OFF_XGF);
  _Float16*  cpad  = (_Float16*)(ws + OFF_CPAD);
  float*     encB  = (float*)(ws + OFF_ENC);

  k_prep<<<1250, 256, 0, stream>>>(comp_w, cg, cbn, cm, cv,
                                   enc_w, eg, ebn, em, ev, ws);
  k_gamma<<<BATCH, 256, 0, stream>>>(x, gamma);
  k_pow<<<4608, 256, 0, stream>>>(x, gamma, xgf, xgh);
  k_comp_gemm<<<288, 256, 0, stream>>>(xgh, compw, cs, cbias, cpad);
  k_enc_gemm<<<72, 256, 0, stream>>>(cpad, encw, esc, ebias, encB);
  k_carafe<<<4608, 256, 0, stream>>>(encB, xgf, out);
}
